// VarInputFeedRNNDecoder_49821620634192
// MI455X (gfx1250) — matrix operand order repaired and checked by reading, not compiled, not run on hardware
//
#include <hip/hip_runtime.h>

// ---------------------------------------------------------------------------
// CDNA5 / gfx1250 wave32 WMMA implementation of the variational input-feed
// LSTM decoder.  All GEMMs run through v_wmma_f32_16x16x32_bf16 with fp32
// accumulation; weights are converted to bf16 once per launch and live in L2
// (46 MB << 192 MB) for the 64 sequential timesteps.
// ---------------------------------------------------------------------------

typedef unsigned short u16;
typedef unsigned int   u32;
typedef __attribute__((ext_vector_type(16))) __bf16 v16bf;
typedef __attribute__((ext_vector_type(8)))  float  v8f;

#define L_T 64
#define L_S 64
#define BATCH 64
#define E_DIM 512
#define H_DIM 1024
#define Z_DIM 128

#define K0 (E_DIM + H_DIM + Z_DIM + H_DIM)  // 2688: [emb|feed|z|h0]
#define K1 (2 * H_DIM)                       // 2048: [h0n|h1prev]
#define N_GATES (4 * H_DIM)                  // 4096

__device__ __forceinline__ u16 f2bf(float f) {
  u32 x = __float_as_uint(f);
  u32 r = x + 0x7FFFu + ((x >> 16) & 1u);   // round-to-nearest-even
  return (u16)(r >> 16);
}
__device__ __forceinline__ float sigm(float x) { return 1.0f / (1.0f + __expf(-x)); }

// ---- WMMA fragment loads (CDNA5 ISA 7.12.2 layouts, wave32) ----------------
// A (16x32 bf16, row-major source): lane L holds row M = L%16.
//   VGPR 0-3: K = (L/16)*8 + 0..7        -> one 16B load
//   VGPR 4-7: K = 16 + (L/16)*8 + 0..7   -> one 16B load
__device__ __forceinline__ v16bf load_a_frag(const u16* __restrict__ A, int lda,
                                             int mBase, int kb) {
  int l = threadIdx.x & 31;
  const u16* p = A + (mBase + (l & 15)) * lda + kb + ((l >> 4) << 3);
  union { uint4 u[2]; v16bf v; } fr;
  fr.u[0] = *(const uint4*)(p);
  fr.u[1] = *(const uint4*)(p + 16);
  return fr.v;
}
// B (32x16 bf16) built from row-major W[N,K]: column n of B = row n of W.
// Lane L holds column N = L%16, K = (L/16)*8 + 0..7 and 16 + (L/16)*8 + 0..7.
__device__ __forceinline__ v16bf load_b_frag(const u16* __restrict__ W, int ldw,
                                             int nBase, int kb) {
  int l = threadIdx.x & 31;
  const u16* p = W + (nBase + (l & 15)) * ldw + kb + ((l >> 4) << 3);
  union { uint4 u[2]; v16bf v; } fr;
  fr.u[0] = *(const uint4*)(p);
  fr.u[1] = *(const uint4*)(p + 16);
  return fr.v;
}

// ---- OUT[64,N] = A[64,K](bf16) @ W[N,K]^T(bf16) + bias, fp32 out -----------
// Block = 128 threads = 4 waves; wave w owns M-tile w; block covers 64 cols.
__global__ void gemm_bf16_wmma(const u16* __restrict__ A, int lda,
                               const u16* __restrict__ W, int ldw,
                               const float* __restrict__ bias,
                               float* __restrict__ OUT, int N, int K) {
  const int wave  = threadIdx.x >> 5;
  const int l     = threadIdx.x & 31;
  const int mBase = wave * 16;
  const int nBlk  = blockIdx.x * 64;

  v8f acc0 = {}, acc1 = {}, acc2 = {}, acc3 = {};
  for (int kb = 0; kb < K; kb += 32) {
    v16bf a  = load_a_frag(A, lda, mBase, kb);
    v16bf b0 = load_b_frag(W, ldw, nBlk + 0,  kb);
    v16bf b1 = load_b_frag(W, ldw, nBlk + 16, kb);
    v16bf b2 = load_b_frag(W, ldw, nBlk + 32, kb);
    v16bf b3 = load_b_frag(W, ldw, nBlk + 48, kb);
    acc0 = __builtin_amdgcn_wmma_f32_16x16x32_bf16(false, a, false, b0, (short)0, acc0, false, false);
    acc1 = __builtin_amdgcn_wmma_f32_16x16x32_bf16(false, a, false, b1, (short)0, acc1, false, false);
    acc2 = __builtin_amdgcn_wmma_f32_16x16x32_bf16(false, a, false, b2, (short)0, acc2, false, false);
    acc3 = __builtin_amdgcn_wmma_f32_16x16x32_bf16(false, a, false, b3, (short)0, acc3, false, false);
  }
  // C/D layout: VGPR r, lanes 0-15: (M=r, N=lane); lanes 16-31: (M=8+r, N=lane-16)
  const int mrow = mBase + ((l >> 4) << 3);
  const int col0 = nBlk + (l & 15);
  v8f accs[4] = {acc0, acc1, acc2, acc3};
#pragma unroll
  for (int nt = 0; nt < 4; ++nt) {
    int col = col0 + nt * 16;
    float bi = bias ? bias[col] : 0.0f;
#pragma unroll
    for (int r = 0; r < 8; ++r)
      OUT[(mrow + r) * N + col] = accs[nt][r] + bi;
  }
}

// ---- one-time (per launch) weight conversion -------------------------------
__global__ void cvt_cat_rows(const float* __restrict__ A, const float* __restrict__ B,
                             u16* __restrict__ dst, int KA, int KB, int rows) {
  int K = KA + KB;
  int total = rows * K;
  int i = blockIdx.x * blockDim.x + threadIdx.x;
  if (i >= total) return;
  int r = i / K, c = i % K;
  float v = (c < KA) ? A[r * KA + c] : B[r * KB + (c - KA)];
  dst[i] = f2bf(v);
}
__global__ void cvt_f2bf(const float* __restrict__ s, u16* __restrict__ d, int n) {
  int i = blockIdx.x * blockDim.x + threadIdx.x;
  if (i < n) d[i] = f2bf(s[i]);
}
__global__ void add_bias(const float* __restrict__ a, const float* __restrict__ b,
                         float* __restrict__ d, int n) {
  int i = blockIdx.x * blockDim.x + threadIdx.x;
  if (i < n) d[i] = a[i] + b[i];
}
__global__ void copy_f32(const float* __restrict__ s, float* __restrict__ d, int n) {
  int i = blockIdx.x * blockDim.x + threadIdx.x;
  if (i < n) d[i] = s[i];
}
__global__ void zero_f32(float* __restrict__ d, int n) {
  int i = blockIdx.x * blockDim.x + threadIdx.x;
  if (i < n) d[i] = 0.0f;
}

// ---- per-step elementwise: z = mu + eps*exp(lv/2), KLD, build [x|h0] -------
__global__ void vae_concat(const float* __restrict__ emb_t, const u16* __restrict__ feed,
                           const u16* __restrict__ h0bf, const float* __restrict__ mulv,
                           const float* __restrict__ eps_t, u16* __restrict__ xh,
                           float* __restrict__ kld) {
  __shared__ float red[256];
  int i = blockIdx.x * 256 + threadIdx.x;
  float kl = 0.0f;
  if (i < BATCH * K0) {
    int b = i / K0, c = i % K0;
    u16 v;
    if (c < E_DIM) {
      v = f2bf(emb_t[b * E_DIM + c]);
    } else if (c < E_DIM + H_DIM) {
      v = feed[b * H_DIM + (c - E_DIM)];
    } else if (c < E_DIM + H_DIM + Z_DIM) {
      int zi = c - (E_DIM + H_DIM);
      float mu = mulv[b * 256 + zi];
      float lv = mulv[b * 256 + 128 + zi];
      float ex = __expf(lv);
      float z  = mu + eps_t[b * Z_DIM + zi] * __expf(0.5f * lv);
      kl = -0.5f * (1.0f + lv - mu * mu - ex);
      v = f2bf(z);
    } else {
      v = h0bf[b * H_DIM + (c - (E_DIM + H_DIM + Z_DIM))];
    }
    xh[i] = v;
  }
  red[threadIdx.x] = kl;
  __syncthreads();
  for (int s = 128; s > 0; s >>= 1) {
    if (threadIdx.x < s) red[threadIdx.x] += red[threadIdx.x + s];
    __syncthreads();
  }
  if (threadIdx.x == 0) atomicAdd(kld, red[0]);
}

// ---- LSTM cell: c' = sig(f)c + sig(i)tanh(g); h' = sig(o)tanh(c') ----------
__global__ void lstm_cell(const float* __restrict__ g, float* __restrict__ c,
                          u16* __restrict__ hbf, u16* __restrict__ cat,
                          int catOff, const u16* __restrict__ hOther) {
  int i = blockIdx.x * blockDim.x + threadIdx.x;
  if (i >= BATCH * H_DIM) return;
  int b = i >> 10, h = i & 1023;
  float gi = g[b * N_GATES + h];
  float gf = g[b * N_GATES + 1024 + h];
  float gg = g[b * N_GATES + 2048 + h];
  float go = g[b * N_GATES + 3072 + h];
  float cn = sigm(gf) * c[i] + sigm(gi) * tanhf(gg);
  float hn = sigm(go) * tanhf(cn);
  c[i] = cn;
  u16 hb = f2bf(hn);
  hbf[i] = hb;
  cat[b * 2048 + catOff + h] = hb;
  if (hOther) cat[b * 2048 + 1024 + h] = hOther[i];  // layer-1's h_prev half
}

// ---- attention: scores -> masked softmax -> ctx (one block per batch b) ----
__global__ void attention(const float* __restrict__ q, const float* __restrict__ enc,
                          const int* __restrict__ lengths, float* __restrict__ attns_t,
                          u16* __restrict__ ctxh) {
  int b = blockIdx.x;
  __shared__ float sc[L_S];
  __shared__ float s_sum;
  int wave = threadIdx.x >> 5, lane = threadIdx.x & 31;
  const float* qb = q + b * H_DIM;
  for (int s = wave; s < L_S; s += 8) {
    const float* m = enc + (s * BATCH + b) * H_DIM;  // encoder_outputs[s,b,:]
    float acc = 0.0f;
    for (int h = lane; h < H_DIM; h += 32) acc += qb[h] * m[h];
    for (int off = 16; off > 0; off >>= 1) acc += __shfl_xor(acc, off, 32);
    if (lane == 0) sc[s] = acc;
  }
  __syncthreads();
  if (threadIdx.x == 0) {
    int len = lengths[b];
    float mx = -3.0e38f;
    for (int s = 0; s < L_S; ++s) {
      float v = (s < len) ? sc[s] : -1.0e30f;
      sc[s] = v;
      mx = v > mx ? v : mx;
    }
    float sum = 0.0f;
    for (int s = 0; s < L_S; ++s) {
      float e = __expf(sc[s] - mx);
      sc[s] = e;
      sum += e;
    }
    s_sum = sum;
  }
  __syncthreads();
  if (threadIdx.x < L_S) {
    float a = sc[threadIdx.x] / s_sum;
    sc[threadIdx.x] = a;
    attns_t[b * L_S + threadIdx.x] = a;
  }
  __syncthreads();
  for (int h = threadIdx.x; h < H_DIM; h += blockDim.x) {
    float acc = 0.0f;
    for (int s = 0; s < L_S; ++s) acc += sc[s] * enc[(s * BATCH + b) * H_DIM + h];
    ctxh[b * 2048 + h] = f2bf(acc);  // first half of [ctx | rnn_out]
  }
}

// ---- attn_h = tanh(out); write dec_outs (fp32) + next feed (bf16) ----------
__global__ void out_tanh(const float* __restrict__ g, float* __restrict__ dec_t,
                         u16* __restrict__ feed) {
  int i = blockIdx.x * blockDim.x + threadIdx.x;
  if (i >= BATCH * H_DIM) return;
  float v = tanhf(g[i]);
  dec_t[i] = v;
  feed[i] = f2bf(v);
}

__global__ void copy1(const float* __restrict__ s, float* __restrict__ d) {
  if (threadIdx.x == 0) d[0] = s[0];
}

// ---------------------------------------------------------------------------
extern "C" void kernel_launch(void* const* d_in, const int* in_sizes, int n_in,
                              void* d_out, int out_size, void* d_ws, size_t ws_size,
                              hipStream_t stream) {
  const float* trg_emb = (const float*)d_in[0];
  const float* enc     = (const float*)d_in[1];
  const int*   lengths = (const int*)d_in[2];
  const float* h0_in   = (const float*)d_in[3];
  const float* c0_in   = (const float*)d_in[4];
  const float* in_feed = (const float*)d_in[5];
  const float* eps     = (const float*)d_in[6];
  const float* W_ih0   = (const float*)d_in[7];
  const float* W_hh0   = (const float*)d_in[8];
  const float* b_ih0   = (const float*)d_in[9];
  const float* b_hh0   = (const float*)d_in[10];
  const float* W_ih1   = (const float*)d_in[11];
  const float* W_hh1   = (const float*)d_in[12];
  const float* b_ih1   = (const float*)d_in[13];
  const float* b_hh1   = (const float*)d_in[14];
  const float* W_mu    = (const float*)d_in[15];
  const float* b_mu    = (const float*)d_in[16];
  const float* W_lv    = (const float*)d_in[17];
  const float* b_lv    = (const float*)d_in[18];
  const float* W_a     = (const float*)d_in[19];
  const float* W_out   = (const float*)d_in[20];
  float* out = (float*)d_out;

  // ---- workspace carve-up (256B aligned) ----
  size_t off = 0;
  char* base = (char*)d_ws;
  auto alloc = [&](size_t bytes) -> void* {
    void* p = base + off;
    off += (bytes + 255) & ~(size_t)255;
    return p;
  };
  u16*   Wcat0 = (u16*)alloc((size_t)N_GATES * K0 * 2);     // [4096,2688] bf16
  u16*   Wcat1 = (u16*)alloc((size_t)N_GATES * K1 * 2);     // [4096,2048]
  u16*   Whead = (u16*)alloc((size_t)256 * H_DIM * 2);      // [256,1024] (mu;lv)
  u16*   Wa    = (u16*)alloc((size_t)H_DIM * H_DIM * 2);    // [1024,1024]
  u16*   Wout  = (u16*)alloc((size_t)H_DIM * 2 * H_DIM * 2);// [1024,2048]
  float* bcat0 = (float*)alloc(N_GATES * 4);
  float* bcat1 = (float*)alloc(N_GATES * 4);
  float* bhead = (float*)alloc(256 * 4);
  u16*   xh    = (u16*)alloc((size_t)BATCH * K0 * 2);
  u16*   x1    = (u16*)alloc((size_t)BATCH * K1 * 2);
  u16*   ctxh  = (u16*)alloc((size_t)BATCH * 2 * H_DIM * 2);
  u16*   h0bf  = (u16*)alloc((size_t)BATCH * H_DIM * 2);
  u16*   h1bf  = (u16*)alloc((size_t)BATCH * H_DIM * 2);
  float* c0s   = (float*)alloc((size_t)BATCH * H_DIM * 4);
  float* c1s   = (float*)alloc((size_t)BATCH * H_DIM * 4);
  u16*   feed  = (u16*)alloc((size_t)BATCH * H_DIM * 2);
  float* gates = (float*)alloc((size_t)BATCH * N_GATES * 4);
  float* mulv  = (float*)alloc((size_t)BATCH * 256 * 4);
  float* qbuf  = (float*)alloc((size_t)BATCH * H_DIM * 4);
  float* kld   = (float*)alloc(256);
  (void)ws_size; (void)n_in; (void)in_sizes; (void)out_size;

  auto nb = [](int n) { return (n + 255) / 256; };

  // ---- one-time conversions: fp32 weights -> bf16 L2-resident copies ----
  cvt_cat_rows<<<nb(N_GATES * K0), 256, 0, stream>>>(W_ih0, W_hh0, Wcat0, E_DIM + H_DIM + Z_DIM, H_DIM, N_GATES);
  cvt_cat_rows<<<nb(N_GATES * K1), 256, 0, stream>>>(W_ih1, W_hh1, Wcat1, H_DIM, H_DIM, N_GATES);
  cvt_f2bf<<<nb(128 * H_DIM), 256, 0, stream>>>(W_mu, Whead, 128 * H_DIM);
  cvt_f2bf<<<nb(128 * H_DIM), 256, 0, stream>>>(W_lv, Whead + 128 * H_DIM, 128 * H_DIM);
  cvt_f2bf<<<nb(H_DIM * H_DIM), 256, 0, stream>>>(W_a, Wa, H_DIM * H_DIM);
  cvt_f2bf<<<nb(H_DIM * 2 * H_DIM), 256, 0, stream>>>(W_out, Wout, H_DIM * 2 * H_DIM);
  add_bias<<<nb(N_GATES), 256, 0, stream>>>(b_ih0, b_hh0, bcat0, N_GATES);
  add_bias<<<nb(N_GATES), 256, 0, stream>>>(b_ih1, b_hh1, bcat1, N_GATES);
  copy_f32<<<1, 256, 0, stream>>>(b_mu, bhead, 128);
  copy_f32<<<1, 256, 0, stream>>>(b_lv, bhead + 128, 128);

  // ---- state init (every call: deterministic across graph replays) ----
  cvt_f2bf<<<nb(BATCH * H_DIM), 256, 0, stream>>>(h0_in, h0bf, BATCH * H_DIM);
  cvt_f2bf<<<nb(BATCH * H_DIM), 256, 0, stream>>>(h0_in + BATCH * H_DIM, h1bf, BATCH * H_DIM);
  copy_f32<<<nb(BATCH * H_DIM), 256, 0, stream>>>(c0_in, c0s, BATCH * H_DIM);
  copy_f32<<<nb(BATCH * H_DIM), 256, 0, stream>>>(c0_in + BATCH * H_DIM, c1s, BATCH * H_DIM);
  cvt_f2bf<<<nb(BATCH * H_DIM), 256, 0, stream>>>(in_feed, feed, BATCH * H_DIM);
  zero_f32<<<1, 32, 0, stream>>>(kld, 1);

  float* dec_outs = out;                                  // [64,64,1024]
  float* attns    = out + (size_t)L_T * BATCH * H_DIM;    // [64,64,64]
  float* kld_out  = attns + (size_t)L_T * BATCH * L_S;    // scalar

  for (int t = 0; t < L_T; ++t) {
    const float* emb_t = trg_emb + (size_t)t * BATCH * E_DIM;
    const float* eps_t = eps + (size_t)t * BATCH * Z_DIM;
    float* attns_t = attns + (size_t)t * BATCH * L_S;
    float* dec_t   = dec_outs + (size_t)t * BATCH * H_DIM;

    // VAE heads: [mu|lv] = h1_prev @ [W_mu;W_lv]^T + [b_mu;b_lv]
    gemm_bf16_wmma<<<256 / 64, 128, 0, stream>>>(h1bf, H_DIM, Whead, H_DIM, bhead, mulv, 256, H_DIM);
    // z, KLD, build xh = [emb | feed | z | h0_prev] (bf16)
    vae_concat<<<nb(BATCH * K0), 256, 0, stream>>>(emb_t, feed, h0bf, mulv, eps_t, xh, kld);
    // layer-0 gates + cell
    gemm_bf16_wmma<<<N_GATES / 64, 128, 0, stream>>>(xh, K0, Wcat0, K0, bcat0, gates, N_GATES, K0);
    lstm_cell<<<nb(BATCH * H_DIM), 256, 0, stream>>>(gates, c0s, h0bf, x1, 0, h1bf);
    // layer-1 gates + cell
    gemm_bf16_wmma<<<N_GATES / 64, 128, 0, stream>>>(x1, K1, Wcat1, K1, bcat1, gates, N_GATES, K1);
    lstm_cell<<<nb(BATCH * H_DIM), 256, 0, stream>>>(gates, c1s, h1bf, ctxh, 1024, nullptr);
    // attention: q = h1 @ W_a^T; scores/softmax/ctx
    gemm_bf16_wmma<<<H_DIM / 64, 128, 0, stream>>>(h1bf, H_DIM, Wa, H_DIM, nullptr, qbuf, H_DIM, H_DIM);
    attention<<<BATCH, 256, 0, stream>>>(qbuf, enc, lengths, attns_t, ctxh);
    // attn_h = tanh([ctx|h1] @ W_out^T); emit dec_outs + next feed
    gemm_bf16_wmma<<<H_DIM / 64, 128, 0, stream>>>(ctxh, 2 * H_DIM, Wout, 2 * H_DIM, nullptr, gates, H_DIM, 2 * H_DIM);
    out_tanh<<<nb(BATCH * H_DIM), 256, 0, stream>>>(gates, dec_t, feed);
  }
  copy1<<<1, 32, 0, stream>>>(kld, kld_out);
}
